// BiLstmCrfModel_48404281426437
// MI455X (gfx1250) — matrix operand order repaired and checked by reading, not compiled, not run on hardware
//
#include <hip/hip_runtime.h>
#include <hip/hip_bf16.h>

// ---------------- problem constants ----------------
#define BB   64
#define SS   512
#define EE   300
#define EP   320           // E padded to multiple of 32
#define HH   256           // hidden per direction
#define G4   1024          // 4*H
#define H2_  512
#define TT   48
#define BSR  (BB * SS)     // 32768 rows

typedef __attribute__((ext_vector_type(16))) _Float16 v16h;
typedef __attribute__((ext_vector_type(8)))  _Float16 v8h;
typedef __attribute__((ext_vector_type(8)))  float    v8f;

union U16 { v16h v; v8h h[2]; };

__device__ __forceinline__ v8f wmma_f16(v16h a, v16h b, v8f c) {
    return __builtin_amdgcn_wmma_f32_16x16x32_f16(false, a, false, b,
                                                  (short)0, c, false, false);
}

// A fragment 16x32 f16, source row-major with leading dim ld (elements).
// lane<16 holds row M=lane, K {0..7,16..23}; lane>=16 row M=lane-16, K {8..15,24..31}
__device__ __forceinline__ v16h load_a(const _Float16* __restrict__ p0, int ld, int lane) {
    int row  = lane & 15;
    int half = lane >> 4;
    const _Float16* p = p0 + (size_t)row * ld + half * 8;
    U16 u;
    u.h[0] = *(const v8h*)(p);
    u.h[1] = *(const v8h*)(p + 16);
    return u.v;
}

// B fragment 32x16 f16: W stored (N,K) row-major with leading dim ld.
// lane<16 holds col N=lane, K {0..7,16..23}; lane>=16 col N=lane-16, K {8..15,24..31}
__device__ __forceinline__ v16h load_b(const _Float16* __restrict__ p0, int ld, int lane) {
    int col  = lane & 15;
    int half = lane >> 4;
    const _Float16* p = p0 + (size_t)col * ld + half * 8;
    U16 u;
    u.h[0] = *(const v8h*)(p);
    u.h[1] = *(const v8h*)(p + 16);
    return u.v;
}

// ---------------- prep kernels ----------------

__global__ void embed_gather(const int* __restrict__ tok,
                             const float* __restrict__ emb,
                             _Float16* __restrict__ x) {
    int idx = blockIdx.x * blockDim.x + threadIdx.x;
    if (idx >= BSR * EP) return;
    int row = idx / EP, col = idx - row * EP;
    float v = 0.f;
    if (col < EE) v = emb[(size_t)tok[row] * EE + col];
    x[idx] = (_Float16)v;
}

__global__ void convert_pad(const float* __restrict__ src, _Float16* __restrict__ dst,
                            int rows, int scols, int dcols) {
    int idx = blockIdx.x * blockDim.x + threadIdx.x;
    if (idx >= rows * dcols) return;
    int r = idx / dcols, c = idx - r * dcols;
    dst[idx] = (_Float16)(c < scols ? src[(size_t)r * scols + c] : 0.f);
}

__global__ void bias_sum(const float* __restrict__ a, const float* __restrict__ b,
                         float* __restrict__ o, int n) {
    int i = blockIdx.x * blockDim.x + threadIdx.x;
    if (i < n) o[i] = a[i] + b[i];
}

// ---------------- input-projection GEMM:  Z = x @ Wih^T + (b_ih + b_hh) ----------------
// one wave -> 32 rows x 64 cols (8 WMMAs per k-step, 12 b128 loads)
// waves_per_eu(2): allow up to ~256 VGPRs so the 8 accumulators + prefetched
// fragments never spill (this kernel has enough per-wave ILP at low occupancy).
__global__ void __launch_bounds__(256)
__attribute__((amdgpu_waves_per_eu(2)))
gemm_zin(const _Float16* __restrict__ A,
         const _Float16* __restrict__ W,
         const float* __restrict__ bias,
         _Float16* __restrict__ Z) {
    int lane = threadIdx.x & 31;
    int wid  = (blockIdx.x * blockDim.x + threadIdx.x) >> 5;
    int rowTile = wid >> 4;          // 0..1023 (32-row tiles)
    int colGrp  = wid & 15;          // 0..15  (64-col groups)
    int m0 = rowTile * 32;
    int n0 = colGrp * 64;
    v8f acc[2][4] = {};
    for (int k0 = 0; k0 < EP; k0 += 32) {
        v16h a0 = load_a(A + (size_t)m0 * EP + k0, EP, lane);
        v16h a1 = load_a(A + (size_t)(m0 + 16) * EP + k0, EP, lane);
#pragma unroll
        for (int j = 0; j < 4; ++j) {
            v16h b = load_b(W + (size_t)(n0 + j * 16) * EP + k0, EP, lane);
            acc[0][j] = wmma_f16(a0, b, acc[0][j]);
            acc[1][j] = wmma_f16(a1, b, acc[1][j]);
        }
    }
    int n = lane & 15, mh = (lane >> 4) * 8;
#pragma unroll
    for (int h = 0; h < 2; ++h) {
#pragma unroll
        for (int j = 0; j < 4; ++j) {
            int c = n0 + j * 16 + n;
            float bz = bias[c];
#pragma unroll
            for (int r = 0; r < 8; ++r)
                Z[(size_t)(m0 + h * 16 + mh + r) * G4 + c] = (_Float16)(acc[h][j][r] + bz);
        }
    }
}

// ---------------- persistent bidirectional LSTM scan ----------------
// grid = 8 blocks: blockIdx.x = dir*4 + batchTile; block = 512 threads = 16 waves.
// wave w owns hidden columns [16w,16w+16); computes all 4 gate tiles (32 WMMAs/step).
// h lives in LDS (double buffered), c lives in VGPRs.
// W_hh fragments for gates 0,1 are hoisted into VGPRs for the whole scan;
// gates 2,3 stream from L2 (whole W_hh is 512 KB/dir -> L2-resident).
// waves_per_eu(4): the 16-wave block occupies exactly 4 waves/SIMD, so give
// each wave the full file/4 VGPR budget (~256) -> the ~215 live VGPRs fit.
__global__ void __launch_bounds__(512)
__attribute__((amdgpu_waves_per_eu(4)))
lstm_scan(const _Float16* __restrict__ Zf, const _Float16* __restrict__ Zb,
          const _Float16* __restrict__ Whf, const _Float16* __restrict__ Whb,
          _Float16* __restrict__ hs) {
    __shared__ _Float16 hbuf[2][16 * HH];
    int dir   = blockIdx.x >> 2;
    int btile = blockIdx.x & 3;
    const _Float16* Z  = dir ? Zb  : Zf;
    const _Float16* Wh = dir ? Whb : Whf;

    int lane = threadIdx.x & 31;
    int w    = threadIdx.x >> 5;       // 0..15
    int n    = lane & 15;
    int half = lane >> 4;
    int mh   = half * 8;
    int col  = w * 16 + n;             // hidden column this lane owns in epilogue

    for (int i = threadIdx.x; i < 16 * HH; i += 512) hbuf[0][i] = (_Float16)0.f;
    v8f c = {};

    // hoist gate-0/1 weight fragments into registers (time-invariant)
    v16h breg[16];
#pragma unroll
    for (int g = 0; g < 2; ++g)
#pragma unroll
        for (int kk = 0; kk < 8; ++kk)
            breg[g * 8 + kk] =
                load_b(Wh + (size_t)(g * HH + w * 16) * HH + kk * 32, HH, lane);

    __syncthreads();

    for (int t = 0; t < SS; ++t) {
        int s = dir ? (SS - 1 - t) : t;
        const _Float16* hp = &hbuf[t & 1][0];
        v8f acc[4] = {};
#pragma unroll
        for (int kk = 0; kk < 8; ++kk) {
            // A fragment from LDS h buffer (row-major 16 x 256)
            U16 ua;
            const _Float16* pa = hp + (size_t)(lane & 15) * HH + kk * 32 + half * 8;
            ua.h[0] = *(const v8h*)(pa);
            ua.h[1] = *(const v8h*)(pa + 16);
            acc[0] = wmma_f16(ua.v, breg[kk], acc[0]);
            acc[1] = wmma_f16(ua.v, breg[8 + kk], acc[1]);
#pragma unroll
            for (int g = 2; g < 4; ++g) {
                v16h b = load_b(Wh + (size_t)(g * HH + w * 16) * HH + kk * 32, HH, lane);
                acc[g] = wmma_f16(ua.v, b, acc[g]);
            }
        }
        _Float16 hn[8];
#pragma unroll
        for (int r = 0; r < 8; ++r) {
            int brow  = btile * 16 + mh + r;          // batch index
            size_t zr = ((size_t)brow * SS + s) * G4;
            float zi = acc[0][r] + (float)Z[zr + 0 * HH + col];
            float zf = acc[1][r] + (float)Z[zr + 1 * HH + col];
            float zg = acc[2][r] + (float)Z[zr + 2 * HH + col];
            float zo = acc[3][r] + (float)Z[zr + 3 * HH + col];
            float ig = 1.f / (1.f + __expf(-zi));
            float fg = 1.f / (1.f + __expf(-zf));
            float gg = tanhf(zg);
            float og = 1.f / (1.f + __expf(-zo));
            float cn = fg * c[r] + ig * gg;
            c[r] = cn;
            float hv = og * tanhf(cn);
            hn[r] = (_Float16)hv;
            hs[((size_t)brow * SS + s) * H2_ + dir * HH + col] = hn[r];
        }
        _Float16* hq = &hbuf[(t + 1) & 1][0];
#pragma unroll
        for (int r = 0; r < 8; ++r) hq[(size_t)(mh + r) * HH + col] = hn[r];
        __syncthreads();
    }
}

// ---------------- classifier GEMM + fused softmax ----------------
// one wave -> 32 rows x 48 cols
__global__ void __launch_bounds__(256)
__attribute__((amdgpu_waves_per_eu(2)))
linsoft(const _Float16* __restrict__ hs,
        const _Float16* __restrict__ Wl,
        const float* __restrict__ bl,
        float* __restrict__ probs) {
    int lane = threadIdx.x & 31;
    int wid  = (blockIdx.x * blockDim.x + threadIdx.x) >> 5;  // 0..1023
    int m0 = wid * 32;
    v8f acc[2][3] = {};
    for (int k0 = 0; k0 < H2_; k0 += 32) {
        v16h a0 = load_a(hs + (size_t)m0 * H2_ + k0, H2_, lane);
        v16h a1 = load_a(hs + (size_t)(m0 + 16) * H2_ + k0, H2_, lane);
#pragma unroll
        for (int j = 0; j < 3; ++j) {
            v16h b = load_b(Wl + (size_t)(j * 16) * H2_ + k0, H2_, lane);
            acc[0][j] = wmma_f16(a0, b, acc[0][j]);
            acc[1][j] = wmma_f16(a1, b, acc[1][j]);
        }
    }
    int n = lane & 15, mh = (lane >> 4) * 8;
    float bv[3];
#pragma unroll
    for (int j = 0; j < 3; ++j) bv[j] = bl[j * 16 + n];
#pragma unroll
    for (int h = 0; h < 2; ++h) {
#pragma unroll
        for (int r = 0; r < 8; ++r) {
            float v0 = acc[h][0][r] + bv[0];
            float v1 = acc[h][1][r] + bv[1];
            float v2 = acc[h][2][r] + bv[2];
            float mx = fmaxf(v0, fmaxf(v1, v2));
#pragma unroll
            for (int d = 8; d >= 1; d >>= 1) mx = fmaxf(mx, __shfl_xor(mx, d, 16));
            float e0 = __expf(v0 - mx), e1 = __expf(v1 - mx), e2 = __expf(v2 - mx);
            float sm = e0 + e1 + e2;
#pragma unroll
            for (int d = 8; d >= 1; d >>= 1) sm += __shfl_xor(sm, d, 16);
            float inv = 1.f / sm;
            size_t row = (size_t)(m0 + h * 16 + mh + r) * TT;
            probs[row + n]      = e0 * inv;
            probs[row + 16 + n] = e1 * inv;
            probs[row + 32 + n] = e2 * inv;
        }
    }
}

// ---------------- CRF loss: one block (64 thr) per batch row ----------------
__global__ void __launch_bounds__(64) crf_scan(
        const float* __restrict__ probs, const int* __restrict__ labels,
        const int* __restrict__ seq_lens, const float* __restrict__ trans,
        const float* __restrict__ startt, const float* __restrict__ endt,
        float* __restrict__ part) {
    __shared__ float sT[TT * TT];
    __shared__ float alpha[TT];
    __shared__ float red[64];
    int b = blockIdx.x;
    int j = threadIdx.x;
    for (int i = j; i < TT * TT; i += 64) sT[i] = trans[i];
    int L = seq_lens[b];
    const float* pb = probs + (size_t)b * SS * TT;
    if (j < TT) alpha[j] = startt[j] + pb[j];
    __syncthreads();
    for (int t = 1; t < SS; ++t) {
        float newv = 0.f, oldv = 0.f;
        if (j < TT) {
            oldv = alpha[j];
            float m = -3.4e38f;
            for (int i = 0; i < TT; ++i) m = fmaxf(m, alpha[i] + sT[i * TT + j]);
            float s = 0.f;
            for (int i = 0; i < TT; ++i) s += __expf(alpha[i] + sT[i * TT + j] - m);
            newv = m + __logf(s) + pb[(size_t)t * TT + j];
        }
        __syncthreads();
        if (j < TT) alpha[j] = (t < L) ? newv : oldv;
        __syncthreads();
    }
    red[j] = (j < TT) ? alpha[j] + endt[j] : -3.4e38f;
    __syncthreads();
    if (j == 0) {
        float m = -3.4e38f;
        for (int i = 0; i < TT; ++i) m = fmaxf(m, red[i]);
        float s = 0.f;
        for (int i = 0; i < TT; ++i) s += __expf(red[i] - m);
        float log_z = m + __logf(s);
        const int* tg = labels + (size_t)b * SS;
        float score = startt[tg[0]];
        for (int t = 0; t + 1 < SS; ++t) {
            if (t + 1 < L) score += sT[tg[t] * TT + tg[t + 1]];
            if (t < L)     score += pb[(size_t)t * TT + tg[t]];
        }
        int last = L - 1;
        score += endt[tg[last]];
        if (SS - 1 < L) score += pb[(size_t)(SS - 1) * TT + tg[SS - 1]];
        part[b] = score - log_z;
    }
}

__global__ void reduce_loss(const float* __restrict__ part, float* __restrict__ out) {
    if (blockIdx.x == 0 && threadIdx.x == 0) {
        float s = 0.f;
        for (int i = 0; i < BB; ++i) s += part[i];   // fixed order -> deterministic
        out[0] = -s;
    }
}

// ---------------- host-side orchestration ----------------
static inline char* ws_take(char*& p, size_t bytes) {
    char* r = p;
    p += (bytes + 255) & ~(size_t)255;
    return r;
}

extern "C" void kernel_launch(void* const* d_in, const int* in_sizes, int n_in,
                              void* d_out, int out_size, void* d_ws, size_t ws_size,
                              hipStream_t stream) {
    (void)in_sizes; (void)n_in; (void)out_size; (void)ws_size;
    const int*   tok     = (const int*)  d_in[0];
    const int*   seqlen  = (const int*)  d_in[1];
    const int*   labels  = (const int*)  d_in[2];
    const float* emb     = (const float*)d_in[3];
    const float* w_ih_f  = (const float*)d_in[4];
    const float* w_hh_f  = (const float*)d_in[5];
    const float* b_ih_f  = (const float*)d_in[6];
    const float* b_hh_f  = (const float*)d_in[7];
    const float* w_ih_b  = (const float*)d_in[8];
    const float* w_hh_b  = (const float*)d_in[9];
    const float* b_ih_b  = (const float*)d_in[10];
    const float* b_hh_b  = (const float*)d_in[11];
    const float* w_lin   = (const float*)d_in[12];
    const float* b_lin   = (const float*)d_in[13];
    const float* trans   = (const float*)d_in[14];
    const float* startt  = (const float*)d_in[15];
    const float* endt    = (const float*)d_in[16];
    float* out = (float*)d_out;

    char* p = (char*)d_ws;
    _Float16* xf16   = (_Float16*)ws_take(p, (size_t)BSR * EP * 2);
    _Float16* wihf_h = (_Float16*)ws_take(p, (size_t)G4 * EP * 2);
    _Float16* wihb_h = (_Float16*)ws_take(p, (size_t)G4 * EP * 2);
    _Float16* whhf_h = (_Float16*)ws_take(p, (size_t)G4 * HH * 2);
    _Float16* whhb_h = (_Float16*)ws_take(p, (size_t)G4 * HH * 2);
    _Float16* wlin_h = (_Float16*)ws_take(p, (size_t)TT * H2_ * 2);
    float*    bzf    = (float*)   ws_take(p, (size_t)G4 * 4);
    float*    bzb    = (float*)   ws_take(p, (size_t)G4 * 4);
    _Float16* Zf     = (_Float16*)ws_take(p, (size_t)BSR * G4 * 2);
    _Float16* Zb     = (_Float16*)ws_take(p, (size_t)BSR * G4 * 2);
    _Float16* hsbuf  = (_Float16*)ws_take(p, (size_t)BSR * H2_ * 2);
    float*    probs  = (float*)   ws_take(p, (size_t)BSR * TT * 4);
    float*    part   = (float*)   ws_take(p, (size_t)BB * 4);

    // 1. embedding gather (f32 -> padded f16)
    embed_gather<<<(BSR * EP + 255) / 256, 256, 0, stream>>>(tok, emb, xf16);

    // 2. weight conversions
    convert_pad<<<(G4 * EP + 255) / 256, 256, 0, stream>>>(w_ih_f, wihf_h, G4, EE, EP);
    convert_pad<<<(G4 * EP + 255) / 256, 256, 0, stream>>>(w_ih_b, wihb_h, G4, EE, EP);
    convert_pad<<<(G4 * HH + 255) / 256, 256, 0, stream>>>(w_hh_f, whhf_h, G4, HH, HH);
    convert_pad<<<(G4 * HH + 255) / 256, 256, 0, stream>>>(w_hh_b, whhb_h, G4, HH, HH);
    convert_pad<<<(TT * H2_ + 255) / 256, 256, 0, stream>>>(w_lin, wlin_h, TT, H2_, H2_);
    bias_sum<<<4, 256, 0, stream>>>(b_ih_f, b_hh_f, bzf, G4);
    bias_sum<<<4, 256, 0, stream>>>(b_ih_b, b_hh_b, bzb, G4);

    // 3. batched input projections (WMMA GEMM, biases fused)
    gemm_zin<<<2048, 256, 0, stream>>>(xf16, wihf_h, bzf, Zf);
    gemm_zin<<<2048, 256, 0, stream>>>(xf16, wihb_h, bzb, Zb);

    // 4. persistent bidirectional LSTM scan (8 workgroups x 512 threads)
    lstm_scan<<<8, 512, 0, stream>>>(Zf, Zb, whhf_h, whhb_h, hsbuf);

    // 5. classifier GEMM + fused softmax (1024 waves -> 128 blocks of 8 waves)
    linsoft<<<128, 256, 0, stream>>>(hsbuf, wlin_h, b_lin, probs);

    // 6. CRF forward scan + gold score, one block per batch row
    crf_scan<<<BB, 64, 0, stream>>>(probs, labels, seqlen, trans, startt, endt, part);

    // 7. deterministic final reduction
    reduce_loss<<<1, 1, 0, stream>>>(part, out);
}
